// StandardAttention_29738353557507
// MI455X (gfx1250) — matrix operand order repaired and checked by reading, not compiled, not run on hardware
//
// CDNA5 (gfx1250) causal MHA: bf16 WMMA everywhere, TDM async A-tile loads,
// flash-style online-softmax attention. Compile-only target: gfx1250.
#include <hip/hip_runtime.h>
#include <stdint.h>

typedef __attribute__((ext_vector_type(16))) __bf16 bf16x16;
typedef __attribute__((ext_vector_type(8)))  float  f32x8;
typedef __attribute__((ext_vector_type(4)))  unsigned int u32x4;
typedef __attribute__((ext_vector_type(8)))  int    i32x8;
typedef __attribute__((ext_vector_type(4)))  int    i32x4;

union FragU { uint4 q[2]; bf16x16 v; };

__device__ inline unsigned short f32_to_bf16(float f) {
  union { float f; uint32_t u; } c; c.f = f;
  uint32_t u = c.u;
  u += 0x7FFFu + ((u >> 16) & 1u);           // round-to-nearest-even
  return (unsigned short)(u >> 16);
}

__device__ inline f32x8 zero8() {
  f32x8 z;
  #pragma unroll
  for (int i = 0; i < 8; i++) z[i] = 0.f;
  return z;
}

__device__ inline f32x8 wmma_bf16(bf16x16 a, bf16x16 b, f32x8 c) {
  // D = A(16x32 bf16) * B(32x16 bf16) + C(16x16 f32)
  return __builtin_amdgcn_wmma_f32_16x16x32_bf16(
      /*neg_a=*/false, a, /*neg_b=*/false, b,
      /*c_mod=*/(short)0, c, /*reuse_a=*/false, /*reuse_b=*/false);
}

// --------------------------------------------------------------- TDM load ---
#if __has_builtin(__builtin_amdgcn_tensor_load_to_lds) && \
    __has_builtin(__builtin_amdgcn_s_wait_tensorcnt)
#define HAVE_TDM 1
#else
#define HAVE_TDM 0
#endif

#if HAVE_TDM
// Async 2-D tile load (row-major, 2-byte elements) into LDS with hardware
// padding of 4 DWORDs every 64 bytes (so a 32-elem row lands on an 80-byte
// = 40-element LDS stride). D# layout per CDNA5 ISA ch.8.
__device__ inline void tdm_load_tile_2d(unsigned lds_off, const void* gaddr,
                                        unsigned tile_d0, unsigned tile_d1,
                                        unsigned tensor_d0, unsigned tensor_d1,
                                        unsigned stride_elems) {
  unsigned long long ga = (unsigned long long)(uintptr_t)gaddr;
  u32x4 g0;
  g0[0] = 1u;                                        // count=1, user mode
  g0[1] = lds_off;                                   // lds_addr (bytes)
  g0[2] = (unsigned)(ga & 0xFFFFFFFFu);              // global_addr[31:0]
  g0[3] = (unsigned)((ga >> 32) & 0x01FFFFFFu)       // global_addr[56:32]
        | (2u << 30);                                // type = 2 ("image")
  i32x8 g1;
  g1[0] = (int)((1u << 16)      // data_size = 2 bytes
              | (1u << 20)      // pad_enable
              | (3u << 22)      // pad_interval: 16 DWORDs (64 B)
              | (3u << 25));    // pad_amount: 4 DWORDs (16 B)
  g1[1] = (int)((tensor_d0 & 0xFFFFu) << 16);        // tensor_dim0[15:0]
  g1[2] = (int)((tensor_d0 >> 16) | ((tensor_d1 & 0xFFFFu) << 16));
  g1[3] = (int)((tensor_d1 >> 16) | (tile_d0 << 16)); // dim1 hi | tile_dim0
  g1[4] = (int)tile_d1;                              // tile_dim1 (tile_dim2=0)
  g1[5] = (int)stride_elems;                         // tensor_dim0_stride lo32
  g1[6] = 0;
  g1[7] = 0;
  i32x4 z4; z4[0] = z4[1] = z4[2] = z4[3] = 0;
#if __clang_major__ >= 23
  i32x8 z8;
  #pragma unroll
  for (int i = 0; i < 8; i++) z8[i] = 0;
  __builtin_amdgcn_tensor_load_to_lds(g0, g1, z4, z4, z8, 0);
#else
  __builtin_amdgcn_tensor_load_to_lds(g0, g1, z4, z4, 0);
#endif
}
#endif

// ---------------------------------------------------------------- convert ---
__global__ void cvt_f32_bf16_x4(const float4* __restrict__ in,
                                ushort4* __restrict__ out, int n4) {
  int i  = blockIdx.x * blockDim.x + threadIdx.x;
  int st = gridDim.x * blockDim.x;
  for (; i < n4; i += st) {
    float4 f = in[i];
    ushort4 o;
    o.x = f32_to_bf16(f.x); o.y = f32_to_bf16(f.y);
    o.z = f32_to_bf16(f.z); o.w = f32_to_bf16(f.w);
    out[i] = o;
  }
}

// ------------------------------------------------------------------- GEMM ---
// C[M,N] = A[M,K] * B[K,N], bf16 row-major inputs, f32 accumulate.
// Block = 256 threads (8 waves). Tile 128x128, K-step 32.
// Wave grid 4x2: wave computes 32 rows (2 m-tiles) x 64 cols (4 n-tiles)
// = 8 WMMAs per K-step. A tile comes in via TDM (wave 0), B tile is
// transposed into LDS by all threads.
#define LSTR 40   // LDS row stride in bf16 elems (80 B, 16B-aligned, padded)

template <bool OUT_BF16>
__global__ __launch_bounds__(256)
void gemm_bf16_wmma(const unsigned short* __restrict__ A,
                    const unsigned short* __restrict__ Bm,
                    void* __restrict__ Cout, int M, int N, int K) {
  __shared__ unsigned short As[128][LSTR];  // A tile, row-major [m][k]
  __shared__ unsigned short Bt[128][LSTR];  // B tile, transposed [n][k]
  const int t = threadIdx.x;
  const int w = t >> 5, l = t & 31;
  const int wr = w >> 1, wc = w & 1;        // wave grid 4 (rows) x 2 (cols)
  const int lhalf = l >> 4, lmod = l & 15;
  const int tilesN = N >> 7;
  const int m0 = (blockIdx.x / tilesN) << 7;
  const int n0 = (blockIdx.x % tilesN) << 7;

  f32x8 acc[2][4];
  #pragma unroll
  for (int mt = 0; mt < 2; mt++)
    #pragma unroll
    for (int tn = 0; tn < 4; tn++) acc[mt][tn] = zero8();

  for (int k0 = 0; k0 < K; k0 += 32) {
    // ---- A tile 128x32 -> LDS -------------------------------------------
#if HAVE_TDM
    if (w == 0) {
      tdm_load_tile_2d((unsigned)(uintptr_t)&As[0][0],
                       A + (size_t)m0 * K + k0,
                       /*tile*/ 32u, 128u,
                       /*tensor*/ (unsigned)K, (unsigned)M,
                       /*stride*/ (unsigned)K);
      __builtin_amdgcn_s_wait_tensorcnt(0);
    }
#else
    {
      const int arow = t >> 1;              // 0..127
      const int acol = (t & 1) << 4;        // 0 or 16
      const uint4* src = (const uint4*)(A + (size_t)(m0 + arow) * K + k0 + acol);
      *(uint4*)&As[arow][acol]     = src[0];
      *(uint4*)&As[arow][acol + 8] = src[1];
    }
#endif
    // ---- B tile 32x128 -> LDS transposed --------------------------------
    #pragma unroll
    for (int i = 0; i < 2; i++) {
      int e  = t * 16 + i * 8;              // element in 32x128 tile
      int kk = e >> 7, nn = e & 127;
      uint4 vv = *(const uint4*)(Bm + (size_t)(k0 + kk) * N + n0 + nn);
      const unsigned short* s = (const unsigned short*)&vv;
      #pragma unroll
      for (int j = 0; j < 8; j++) Bt[nn + j][kk] = s[j];
    }
    __syncthreads();

    FragU fa[2];  // A frag: lane l -> M=l%16, K=(l/16)*8 + {0..7, 16..23}
    #pragma unroll
    for (int mt = 0; mt < 2; mt++) {
      int row = (wr << 5) + (mt << 4) + lmod, kb = lhalf << 3;
      fa[mt].q[0] = *(const uint4*)&As[row][kb];
      fa[mt].q[1] = *(const uint4*)&As[row][kb + 16];
    }
    #pragma unroll
    for (int tn = 0; tn < 4; tn++) {
      FragU fb;  // B frag: lane l -> N=l%16, K=(l/16)*8 + {0..7, 16..23}
      int n = (wc << 6) + (tn << 4) + lmod, kb = lhalf << 3;
      fb.q[0] = *(const uint4*)&Bt[n][kb];
      fb.q[1] = *(const uint4*)&Bt[n][kb + 16];
      #pragma unroll
      for (int mt = 0; mt < 2; mt++)
        acc[mt][tn] = wmma_bf16(fa[mt].v, fb.v, acc[mt][tn]);
    }
    __syncthreads();
  }
  // Store: C/D layout -> lane l holds (M=(l/16)*8+r, N=l%16)
  #pragma unroll
  for (int mt = 0; mt < 2; mt++) {
    #pragma unroll
    for (int tn = 0; tn < 4; tn++) {
      #pragma unroll
      for (int r = 0; r < 8; r++) {
        int row = m0 + (wr << 5) + (mt << 4) + (lhalf << 3) + r;
        int col = n0 + (wc << 6) + (tn << 4) + lmod;
        float v = acc[mt][tn][r];
        if (OUT_BF16) ((unsigned short*)Cout)[(size_t)row * N + col] = f32_to_bf16(v);
        else          ((float*)Cout)[(size_t)row * N + col] = v;
      }
    }
  }
}

// -------------------------------------------------------------- attention ---
#define SEQ 2048
#define NH  16
#define HD  64

__global__ __launch_bounds__(128)
void attn_fwd_wmma(const unsigned short* __restrict__ Q,
                   const unsigned short* __restrict__ Kb,
                   const unsigned short* __restrict__ Vb,
                   unsigned short* __restrict__ O) {
  __shared__ unsigned short Vt[64][LSTR];       // V block transposed [n][k]
  __shared__ unsigned short Pt[4][16][LSTR];    // per-wave P scratch
  const int t = threadIdx.x, w = t >> 5, l = t & 31;
  const int lhalf = l >> 4, lmod = l & 15;
  const int qblocks = SEQ / 64;
  const int bid = blockIdx.x;
  const int b  = bid / (NH * qblocks);
  const int h  = (bid / qblocks) % NH;
  const int q0 = (bid % qblocks) << 6;
  const size_t rstr = (size_t)NH * HD;              // 1024
  const size_t base = (size_t)b * SEQ * rstr + (size_t)h * HD;

  // Q A-fragments for both halves of head dim (direct from global)
  FragU fq[2];
  {
    const unsigned short* qp = Q + base + (size_t)(q0 + (w << 4) + lmod) * rstr;
    #pragma unroll
    for (int hf = 0; hf < 2; hf++) {
      int kb = hf * 32 + (lhalf << 3);
      fq[hf].q[0] = *(const uint4*)(qp + kb);
      fq[hf].q[1] = *(const uint4*)(qp + kb + 16);
    }
  }

  f32x8 acc[4];
  #pragma unroll
  for (int i = 0; i < 4; i++) acc[i] = zero8();
  float mrow[8], lrow[8];
  #pragma unroll
  for (int r = 0; r < 8; r++) { mrow[r] = -3.0e38f; lrow[r] = 0.f; }

  const int kend = q0 + 64;                      // causal extent of this block
  for (int kblk = 0; kblk < kend; kblk += 32) {
    __syncthreads();
    // Stage V^T (32 keys x 64 cols) cooperatively
    #pragma unroll
    for (int i = 0; i < 2; i++) {
      int e = t * 16 + i * 8;
      int kk = e >> 6, nn = e & 63;
      uint4 vv = *(const uint4*)(Vb + base + (size_t)(kblk + kk) * rstr + nn);
      const unsigned short* s = (const unsigned short*)&vv;
      #pragma unroll
      for (int j = 0; j < 8; j++) Vt[nn + j][kk] = s[j];
    }
    // Scores S = Q K^T (K^T B-frags are contiguous runs of K's rows in global)
    f32x8 sc[2];
    #pragma unroll
    for (int kt = 0; kt < 2; kt++) {
      sc[kt] = zero8();
      const unsigned short* kp =
          Kb + base + (size_t)(kblk + (kt << 4) + lmod) * rstr;
      #pragma unroll
      for (int hf = 0; hf < 2; hf++) {
        FragU fk;
        int kd = hf * 32 + (lhalf << 3);
        fk.q[0] = *(const uint4*)(kp + kd);
        fk.q[1] = *(const uint4*)(kp + kd + 16);
        sc[kt] = wmma_bf16(fq[hf].v, fk.v, sc[kt]);
      }
    }
    // Causal mask + online softmax (row r lives in a 16-lane half)
    float pv[2][8];
    #pragma unroll
    for (int r = 0; r < 8; r++) {
      int qr = q0 + (w << 4) + (lhalf << 3) + r;
      float s0 = sc[0][r] * 0.125f;              // 1/sqrt(64)
      float s1 = sc[1][r] * 0.125f;
      if (kblk + lmod      > qr) s0 = -3.0e38f;
      if (kblk + 16 + lmod > qr) s1 = -3.0e38f;
      float rm = fmaxf(s0, s1);
      #pragma unroll
      for (int off = 1; off < 16; off <<= 1)
        rm = fmaxf(rm, __shfl_xor(rm, off, 32));
      float mnew  = fmaxf(mrow[r], rm);
      float alpha = __expf(mrow[r] - mnew);
      float p0 = __expf(s0 - mnew);
      float p1 = __expf(s1 - mnew);
      float rs = p0 + p1;
      #pragma unroll
      for (int off = 1; off < 16; off <<= 1)
        rs += __shfl_xor(rs, off, 32);
      lrow[r] = lrow[r] * alpha + rs;
      mrow[r] = mnew;
      #pragma unroll
      for (int tn = 0; tn < 4; tn++) acc[tn][r] *= alpha;
      pv[0][r] = p0; pv[1][r] = p1;
    }
    // Re-layout P: C-format f32 -> A-format bf16 via per-wave LDS scratch
    #pragma unroll
    for (int kt = 0; kt < 2; kt++)
      #pragma unroll
      for (int r = 0; r < 8; r++)
        Pt[w][(lhalf << 3) + r][(kt << 4) + lmod] = f32_to_bf16(pv[kt][r]);
    __syncthreads();
    // acc += P(16x32) @ V(32x64)
    FragU fp;
    {
      int kb = lhalf << 3;
      fp.q[0] = *(const uint4*)&Pt[w][lmod][kb];
      fp.q[1] = *(const uint4*)&Pt[w][lmod][kb + 16];
    }
    #pragma unroll
    for (int tn = 0; tn < 4; tn++) {
      FragU fv;
      int n = (tn << 4) + lmod, kb = lhalf << 3;
      fv.q[0] = *(const uint4*)&Vt[n][kb];
      fv.q[1] = *(const uint4*)&Vt[n][kb + 16];
      acc[tn] = wmma_bf16(fp.v, fv.v, acc[tn]);
    }
  }
  // Epilogue: normalize and store bf16
  #pragma unroll
  for (int tn = 0; tn < 4; tn++) {
    #pragma unroll
    for (int r = 0; r < 8; r++) {
      int qr  = q0 + (w << 4) + (lhalf << 3) + r;
      int col = (tn << 4) + lmod;
      O[base + (size_t)qr * rstr + col] = f32_to_bf16(acc[tn][r] / lrow[r]);
    }
  }
}

// ----------------------------------------------------------------- launch ---
extern "C" void kernel_launch(void* const* d_in, const int* in_sizes, int n_in,
                              void* d_out, int out_size, void* d_ws, size_t ws_size,
                              hipStream_t stream) {
  const float* x  = (const float*)d_in[0];
  const float* Wq = (const float*)d_in[1];
  const float* Wk = (const float*)d_in[2];
  const float* Wv = (const float*)d_in[3];
  const float* Wo = (const float*)d_in[4];

  const int B = 2, S = SEQ, D = 1024;
  const size_t nx = (size_t)B * S * D;   // 4,194,304
  const size_t nw = (size_t)D * D;       // 1,048,576

  unsigned short* ws   = (unsigned short*)d_ws;
  unsigned short* x16  = ws;
  unsigned short* Wq16 = x16 + nx;
  unsigned short* Wk16 = Wq16 + nw;
  unsigned short* Wv16 = Wk16 + nw;
  unsigned short* Wo16 = Wv16 + nw;
  unsigned short* Q16  = Wo16 + nw;
  unsigned short* K16  = Q16 + nx;
  unsigned short* V16  = K16 + nx;
  unsigned short* AO16 = V16 + nx;       // total ~50 MB of d_ws

  // 1) fp32 -> bf16
  cvt_f32_bf16_x4<<<2048, 256, 0, stream>>>((const float4*)x,  (ushort4*)x16,  (int)(nx / 4));
  cvt_f32_bf16_x4<<<1024, 256, 0, stream>>>((const float4*)Wq, (ushort4*)Wq16, (int)(nw / 4));
  cvt_f32_bf16_x4<<<1024, 256, 0, stream>>>((const float4*)Wk, (ushort4*)Wk16, (int)(nw / 4));
  cvt_f32_bf16_x4<<<1024, 256, 0, stream>>>((const float4*)Wv, (ushort4*)Wv16, (int)(nw / 4));
  cvt_f32_bf16_x4<<<1024, 256, 0, stream>>>((const float4*)Wo, (ushort4*)Wo16, (int)(nw / 4));

  // 2) Q/K/V projections (bf16 out)
  const int M = B * S;                    // 4096
  dim3 gg((M / 128) * (D / 128));         // 256 blocks x 256 threads
  gemm_bf16_wmma<true><<<gg, 256, 0, stream>>>(x16, Wq16, Q16, M, D, D);
  gemm_bf16_wmma<true><<<gg, 256, 0, stream>>>(x16, Wk16, K16, M, D, D);
  gemm_bf16_wmma<true><<<gg, 256, 0, stream>>>(x16, Wv16, V16, M, D, D);

  // 3) causal flash attention
  attn_fwd_wmma<<<B * NH * (S / 64), 128, 0, stream>>>(Q16, K16, V16, AO16);

  // 4) output projection (f32 out -> d_out)
  gemm_bf16_wmma<false><<<gg, 256, 0, stream>>>(AO16, Wo16, d_out, M, D, D);
}
